// PermutationMatrix_9912784519795
// MI455X (gfx1250) — matrix operand order repaired and checked by reading, not compiled, not run on hardware
//
#include <hip/hip_runtime.h>
#include <hip/hip_bf16.h>
#include <math.h>

// ---------------------------------------------------------------------------
// PermutationMatrix pipeline for MI455X (gfx1250, wave32, WMMA)
//
//   1) M0 = exp(T*(matrix - rowmax))                         [N^2 elementwise]
//   2) Sinkhorn (20 it) via scaling vectors:  r=1/(M0 c), c=1/(M0^T r)
//      -> 40 L2-resident matvecs instead of 40 full-matrix rewrites
//   3) m = diag(r) M0 diag(c);  B = s * row-cumsum(m)   (L^T = s*triu(ones))
//   4) out = B @ m^T  -- single 4096^3 NT-GEMM using bf16x3 split WMMA
//      (v_wmma_f32_16x16x32_bf16: hi*hi + hi*lo + lo*hi  ~= fp32 accuracy)
//      with double-buffered LDS to overlap L2 traffic with the matrix core.
// ---------------------------------------------------------------------------

#define N4K 4096

typedef __attribute__((ext_vector_type(16))) __bf16 v16bf;
typedef __attribute__((ext_vector_type(8)))  __bf16 v8bf;
typedef __attribute__((ext_vector_type(8)))  float  v8f;

union FragBF { v16bf v; v8bf h[2]; };

// float -> bf16 (RNE) via bit ops (no reliance on __bf16 scalar conversion)
static __device__ __forceinline__ unsigned short f2bf(float f) {
  unsigned u = __float_as_uint(f);
  unsigned r = (u + 0x7FFFu + ((u >> 16) & 1u)) >> 16;
  return (unsigned short)r;
}
static __device__ __forceinline__ float bf2f(unsigned short h) {
  return __uint_as_float(((unsigned)h) << 16);
}

// ---------------------------------------------------------------------------
// 1) per-row max + exp, temperature derived from epoch scalar on device
// ---------------------------------------------------------------------------
__global__ __launch_bounds__(256) void k_rowmax_exp(const float* __restrict__ mat,
                                                    const int* __restrict__ epoch,
                                                    float* __restrict__ M0) {
  __shared__ float sh[256];
  const int row = blockIdx.x;
  const int t = threadIdx.x;
  const float* src = mat + (size_t)row * N4K;
  float vals[16];
  float mx = -3.402823466e38f;
#pragma unroll
  for (int u = 0; u < 16; ++u) {
    float x = src[u * 256 + t];
    vals[u] = x;
    mx = fmaxf(mx, x);
  }
  sh[t] = mx;
  __syncthreads();
  for (int off = 128; off > 0; off >>= 1) {
    if (t < off) sh[t] = fmaxf(sh[t], sh[t + off]);
    __syncthreads();
  }
  const float rowmax = sh[0];
  const float temp = (float)(epoch[0] / 10 + 1) * 0.5f;
  float* dst = M0 + (size_t)row * N4K;
#pragma unroll
  for (int u = 0; u < 16; ++u) dst[u * 256 + t] = expf(temp * (vals[u] - rowmax));
}

// ---------------------------------------------------------------------------
// small utility kernels
// ---------------------------------------------------------------------------
__global__ __launch_bounds__(256) void k_fill(float* __restrict__ p, float val, int n) {
  int i = blockIdx.x * 256 + threadIdx.x;
  if (i < n) p[i] = val;
}

__global__ __launch_bounds__(256) void k_recip(const float* __restrict__ v,
                                               float* __restrict__ c, int n) {
  int i = blockIdx.x * 256 + threadIdx.x;
  if (i < n) c[i] = 1.0f / v[i];
}

// r[i] = 1 / dot(M0[i,:], c)   -- one block per row, coalesced row reads
__global__ __launch_bounds__(256) void k_rowdot_recip(const float* __restrict__ M0,
                                                      const float* __restrict__ c,
                                                      float* __restrict__ r) {
  __shared__ float sh[256];
  const int row = blockIdx.x;
  const int t = threadIdx.x;
  const float* src = M0 + (size_t)row * N4K;
  float s = 0.0f;
#pragma unroll
  for (int u = 0; u < 16; ++u) {
    int j = u * 256 + t;
    s += src[j] * c[j];
  }
  sh[t] = s;
  __syncthreads();
  for (int off = 128; off > 0; off >>= 1) {
    if (t < off) sh[t] += sh[t + off];
    __syncthreads();
  }
  if (t == 0) r[row] = 1.0f / sh[0];
}

// v[col] += sum_{i in 64-row band} r[i]*M0[i,col]  -- coalesced, atomic combine
__global__ __launch_bounds__(256) void k_coldot_atomic(const float* __restrict__ M0,
                                                       const float* __restrict__ r,
                                                       float* __restrict__ v) {
  const int col = blockIdx.x * 256 + threadIdx.x;
  const int i0 = blockIdx.y * 64;
  float s = 0.0f;
#pragma unroll 8
  for (int u = 0; u < 64; ++u) {
    s += r[i0 + u] * M0[(size_t)(i0 + u) * N4K + col];
  }
  atomicAdd(&v[col], s);
}

// ---------------------------------------------------------------------------
// 3) m = r_i * M0 * c_j (in-place over M0), B = s * inclusive row-cumsum(m)
// ---------------------------------------------------------------------------
__global__ __launch_bounds__(256) void k_scale_cumsum(float* __restrict__ M,
                                                      const float* __restrict__ r,
                                                      const float* __restrict__ c,
                                                      const float* __restrict__ lower,
                                                      float* __restrict__ B) {
  __shared__ float sh[256];
  const int row = blockIdx.x;
  const int t = threadIdx.x;
  const float ri = r[row];
  const float s = 1.0f / (1.0f + expf(-5.0f * lower[0]));  // sigmoid(5*lower)
  float* mrow = M + (size_t)row * N4K;
  const int j0 = t * 16;
  float local[16];
  float run = 0.0f;
#pragma unroll
  for (int u = 0; u < 16; ++u) {
    float mv = ri * mrow[j0 + u] * c[j0 + u];
    mrow[j0 + u] = mv;  // M0 -> m (in place; block owns this row)
    run += mv;
    local[u] = run;
  }
  sh[t] = run;
  __syncthreads();
  // inclusive Hillis-Steele scan over 256 thread totals
  for (int off = 1; off < 256; off <<= 1) {
    float val = (t >= off) ? sh[t - off] : 0.0f;
    __syncthreads();
    sh[t] += val;
    __syncthreads();
  }
  const float excl = sh[t] - run;
  float* brow = B + (size_t)row * N4K;
#pragma unroll
  for (int u = 0; u < 16; ++u) brow[j0 + u] = s * (excl + local[u]);
}

// ---------------------------------------------------------------------------
// 4) out[i,k] = sum_j A[i,j] * Bm[k,j]   (NT GEMM, bf16x3 split WMMA)
//    128x128 C tile / block, 8 waves each own 32x64, K stepped by 32,
//    double-buffered LDS (global loads for tile t+1 issued before compute t).
// ---------------------------------------------------------------------------
#define TILE 128
#define KSTEP 32
#define LSTR 40  // padded LDS row stride (bf16 elems): 80B, 16B aligned, conflict-free

__global__ __launch_bounds__(256) void k_gemm_bt(const float* __restrict__ A,   // B-cumsum [N][N]
                                                 const float* __restrict__ Bm,  // m        [N][N]
                                                 float* __restrict__ out) {
  __shared__ unsigned short sAhi[2][TILE * LSTR];
  __shared__ unsigned short sAlo[2][TILE * LSTR];
  __shared__ unsigned short sBhi[2][TILE * LSTR];
  __shared__ unsigned short sBlo[2][TILE * LSTR];

  const int t = threadIdx.x;
  const int lane = t & 31;
  const int wave = t >> 5;
  const int wm = wave & 3;   // row offset *32 within tile
  const int wn = wave >> 2;  // col offset *64 within tile
  const int l15 = lane & 15;
  const bool hi16 = (lane & 16) != 0;

  const int rowBase = blockIdx.y * TILE;
  const int colBase = blockIdx.x * TILE;

  v8f acc[2][4];
#pragma unroll
  for (int i = 0; i < 2; ++i)
#pragma unroll
    for (int j = 0; j < 4; ++j) {
      v8f z = {0.f, 0.f, 0.f, 0.f, 0.f, 0.f, 0.f, 0.f};
      acc[i][j] = z;
    }

  const int ldRow = t >> 1;        // 0..127
  const int ldCol = (t & 1) * 16;  // 0 or 16
  const float* gA = A + (size_t)(rowBase + ldRow) * N4K + ldCol;
  const float* gB = Bm + (size_t)(colBase + ldRow) * N4K + ldCol;
  const int base = ldRow * LSTR + ldCol;

  float4 ra[4], rb[4];

  // ---- preload K-tile 0 into registers, convert+stage into LDS buffer 0
#pragma unroll
  for (int q = 0; q < 4; ++q) {
    ra[q] = *(const float4*)(gA + q * 4);
    rb[q] = *(const float4*)(gB + q * 4);
  }
#pragma unroll
  for (int q = 0; q < 4; ++q) {
    const float ax[4] = {ra[q].x, ra[q].y, ra[q].z, ra[q].w};
    const float bx[4] = {rb[q].x, rb[q].y, rb[q].z, rb[q].w};
#pragma unroll
    for (int e = 0; e < 4; ++e) {
      unsigned short h = f2bf(ax[e]);
      sAhi[0][base + q * 4 + e] = h;
      sAlo[0][base + q * 4 + e] = f2bf(ax[e] - bf2f(h));
      unsigned short g = f2bf(bx[e]);
      sBhi[0][base + q * 4 + e] = g;
      sBlo[0][base + q * 4 + e] = f2bf(bx[e] - bf2f(g));
    }
  }
  __syncthreads();

  const int kA0 = hi16 ? 8 : 0;   // A fragment K origin (16-bit A 16x32 layout)
  const int kB0 = hi16 ? 8 : 0;   // B fragment K origin (same k order as A)

  int cur = 0;
  for (int kk = 0; kk < N4K; kk += KSTEP) {
    const int nxt = cur ^ 1;
    const bool hasNext = (kk + KSTEP) < N4K;

    // ---- issue global loads for the NEXT tile first (overlap with WMMA)
    if (hasNext) {
#pragma unroll
      for (int q = 0; q < 4; ++q) {
        ra[q] = *(const float4*)(gA + kk + KSTEP + q * 4);
        rb[q] = *(const float4*)(gB + kk + KSTEP + q * 4);
      }
    }

    // ---- compute from buffer `cur`: per wave 2x4 subtiles, 3 WMMAs each
#pragma unroll
    for (int tm = 0; tm < 2; ++tm) {
      const int ar = (wm * 32 + tm * 16 + l15) * LSTR;
      FragBF ahi, alo;
      ahi.h[0] = *(const v8bf*)&sAhi[cur][ar + kA0];
      ahi.h[1] = *(const v8bf*)&sAhi[cur][ar + kA0 + 16];
      alo.h[0] = *(const v8bf*)&sAlo[cur][ar + kA0];
      alo.h[1] = *(const v8bf*)&sAlo[cur][ar + kA0 + 16];
#pragma unroll
      for (int tn = 0; tn < 4; ++tn) {
        const int bn = (wn * 64 + tn * 16 + l15) * LSTR;
        FragBF bhi, blo;
        bhi.h[0] = *(const v8bf*)&sBhi[cur][bn + kB0];
        bhi.h[1] = *(const v8bf*)&sBhi[cur][bn + kB0 + 16];
        blo.h[0] = *(const v8bf*)&sBlo[cur][bn + kB0];
        blo.h[1] = *(const v8bf*)&sBlo[cur][bn + kB0 + 16];

        acc[tm][tn] = __builtin_amdgcn_wmma_f32_16x16x32_bf16(
            false, ahi.v, false, bhi.v, (short)0, acc[tm][tn], false, false);
        acc[tm][tn] = __builtin_amdgcn_wmma_f32_16x16x32_bf16(
            false, ahi.v, false, blo.v, (short)0, acc[tm][tn], false, false);
        acc[tm][tn] = __builtin_amdgcn_wmma_f32_16x16x32_bf16(
            false, alo.v, false, bhi.v, (short)0, acc[tm][tn], false, false);
      }
    }

    // ---- convert + stage next tile into the other buffer, single barrier
    if (hasNext) {
#pragma unroll
      for (int q = 0; q < 4; ++q) {
        const float ax[4] = {ra[q].x, ra[q].y, ra[q].z, ra[q].w};
        const float bx[4] = {rb[q].x, rb[q].y, rb[q].z, rb[q].w};
#pragma unroll
        for (int e = 0; e < 4; ++e) {
          unsigned short h = f2bf(ax[e]);
          sAhi[nxt][base + q * 4 + e] = h;
          sAlo[nxt][base + q * 4 + e] = f2bf(ax[e] - bf2f(h));
          unsigned short g = f2bf(bx[e]);
          sBhi[nxt][base + q * 4 + e] = g;
          sBlo[nxt][base + q * 4 + e] = f2bf(bx[e] - bf2f(g));
        }
      }
      __syncthreads();
    }
    cur = nxt;
  }

  // ---- write C: lane row-split per 16x16 f32 D layout (lanes>=16 -> M+8)
  const int rbase = rowBase + wm * 32 + (hi16 ? 8 : 0);
  const int cbase = colBase + wn * 64 + l15;
#pragma unroll
  for (int tm = 0; tm < 2; ++tm)
#pragma unroll
    for (int tn = 0; tn < 4; ++tn)
#pragma unroll
      for (int rr = 0; rr < 8; ++rr)
        out[(size_t)(rbase + tm * 16 + rr) * N4K + cbase + tn * 16] = acc[tm][tn][rr];
}

// ---------------------------------------------------------------------------
// host-side orchestration (graph-capture safe: kernels only, all on `stream`)
// ---------------------------------------------------------------------------
extern "C" void kernel_launch(void* const* d_in, const int* in_sizes, int n_in,
                              void* d_out, int out_size, void* d_ws, size_t ws_size,
                              hipStream_t stream) {
  const float* matrix = (const float*)d_in[0];
  const float* lower = (const float*)d_in[1];
  const int* epoch = (const int*)d_in[2];
  float* out = (float*)d_out;

  char* ws = (char*)d_ws;
  float* M0 = (float*)ws;                                  // 64 MB, becomes m
  float* Bc = (float*)(ws + (size_t)N4K * N4K * 4);        // 64 MB, cumsum B
  float* rv = (float*)(ws + (size_t)2 * N4K * N4K * 4);    // N
  float* vv = rv + N4K;                                    // N
  float* cv = rv + 2 * N4K;                                // N

  // 1) stabilized exp
  k_rowmax_exp<<<N4K, 256, 0, stream>>>(matrix, epoch, M0);

  // 2) Sinkhorn via scaling vectors (20 iterations; epoch=5 -> factor 1)
  k_fill<<<16, 256, 0, stream>>>(cv, 1.0f, N4K);
  for (int it = 0; it < 20; ++it) {
    k_rowdot_recip<<<N4K, 256, 0, stream>>>(M0, cv, rv);
    k_fill<<<16, 256, 0, stream>>>(vv, 0.0f, N4K);
    k_coldot_atomic<<<dim3(16, 64), 256, 0, stream>>>(M0, rv, vv);
    k_recip<<<16, 256, 0, stream>>>(vv, cv, N4K);
  }

  // 3) materialize m (in place) + B = s*cumsum(m)
  k_scale_cumsum<<<N4K, 256, 0, stream>>>(M0, rv, cv, lower, Bc);

  // 4) out = B @ m^T  (bf16x3 WMMA GEMM, double-buffered LDS)
  k_gemm_bt<<<dim3(N4K / TILE, N4K / TILE), 256, 0, stream>>>(Bc, M0, out);
}
